// LayoutLMv3Encoder_86517821210717
// MI455X (gfx1250) — matrix operand order repaired and checked by reading, not compiled, not run on hardware
//
#include <hip/hip_runtime.h>
#include <cmath>

// ---------------------------------------------------------------------------
// LayoutLMv3 encoder forward for gfx1250 (MI455X).
// - All GEMM operands bf16 (weights pre-transposed to [N][K] once per fwd)
// - GEMM core: v_wmma_f32_16x16x32_bf16, double-buffered LDS tiles
// - Tile staging via Tensor Data Mover (tensor_load_to_lds) with D# padding
//   reproducing the conflict-free LDS pitch; completion via TENSORcnt.
// Dims fixed by reference: B=4, S=512, D=768, H=12, DH=64, FF=3072, L=12.
// ---------------------------------------------------------------------------

#define DEV __device__ __forceinline__

typedef __attribute__((ext_vector_type(16))) __bf16          v16bf;
typedef __attribute__((ext_vector_type(8)))  float           v8f;
typedef __attribute__((ext_vector_type(8)))  unsigned short  u16x8;
typedef __attribute__((ext_vector_type(4)))  unsigned int    u32x4;
typedef __attribute__((ext_vector_type(8)))  int             i32x8;
typedef __attribute__((ext_vector_type(4)))  int             i32x4;
typedef unsigned short us;
typedef long long      ll;

static constexpr int Lnum = 12;
static constexpr int Dm   = 768;
static constexpr int Hh   = 12;
static constexpr int FFd  = 3072;
static constexpr int DHd  = 64;
static constexpr int Bb   = 4;
static constexpr int Ss   = 512;
static constexpr int Mrows = Bb * Ss;   // 2048

// fp32 -> bf16 round-to-nearest-even
DEV us f2bf(float x) {
    unsigned u = __builtin_bit_cast(unsigned, x);
    unsigned r = u + 0x7FFFu + ((u >> 16) & 1u);
    return (us)(r >> 16);
}

union FragU { us s[16]; u16x8 h[2]; v16bf v; };

// A-fragment (16x32 bf16): lane<16 -> row=lane, k={0..7,16..23};
// lane>=16 -> row=lane-16, k={8..15,24..31}. p = row base + khalf.
DEV v16bf load_frag_A(const us* p) {
    FragU f;
    f.h[0] = *(const u16x8*)(p);
    f.h[1] = *(const u16x8*)(p + 16);
    return f.v;
}
// B-fragment (32x16 bf16 from Bt[n][k]): lane<16 -> col=lane, k={0..7,16..23};
// lane>=16 -> col=lane-16, k={8..15,24..31}. p = col base + khalf.
DEV v16bf load_frag_B(const us* p) {
    FragU f;
    f.h[0] = *(const u16x8*)(p);
    f.h[1] = *(const u16x8*)(p + 16);
    return f.v;
}

DEV float wave_sum(float x) {
    #pragma unroll
    for (int m = 16; m > 0; m >>= 1) x += __shfl_xor(x, m, 32);
    return x;
}
DEV float wave_max(float x) {
    #pragma unroll
    for (int m = 16; m > 0; m >>= 1) x = fmaxf(x, __shfl_xor(x, m, 32));
    return x;
}

// ---------------------------------------------------------------------------
// TDM: load a [tileRows x 32] bf16 tile (row pitch = pitchElems bf16) from
// global to LDS, depositing rows at a 40-u16 pitch (64B row + 16B pad) via
// D# pad fields: pad_interval=3 (every 16 DWORDs), pad_amount=3 (4 DWORDs).
// D# packing per CDNA5 ISA (2D, groups 2/3 = 0).
// ---------------------------------------------------------------------------
DEV void tdm_load_tile_2d(unsigned ldsOff, const us* gsrc, int tileRows, int pitchElems)
{
    const unsigned long long ga = (unsigned long long)(size_t)gsrc;
    u32x4 g0;
    g0[0] = 1u;                                              // count=1, user mode
    g0[1] = ldsOff;                                          // lds_addr (bytes)
    g0[2] = (unsigned)(ga & 0xFFFFFFFFull);                  // global_addr[31:0]
    g0[3] = (unsigned)((ga >> 32) & 0x1FFFFFFull)            // global_addr[56:32]
          | (2u << 30);                                      // type=2 ("image")
    const unsigned long long s0 = (unsigned long long)pitchElems;
    i32x8 g1;
    g1[0] = (int)((1u << 16)                                 // data_size=2B
                | (1u << 20)                                 // pad_enable
                | (3u << 22)                                 // pad_interval: 16 DW
                | (3u << 25));                               // pad_amount: 4 DW
    g1[1] = (int)(32u << 16);                                // tensor_dim0 = 32
    g1[2] = (int)(((unsigned)tileRows & 0xFFFFu) << 16);     // tensor_dim1 = rows
    g1[3] = (int)(32u << 16);                                // tile_dim0 = 32
    g1[4] = (int)((unsigned)tileRows & 0xFFFFu);             // tile_dim1 = rows
    g1[5] = (int)(s0 & 0xFFFFFFFFull);                       // dim0_stride lo32
    g1[6] = (int)((s0 >> 32) & 0xFFFFull);                   // dim0_stride hi16
    g1[7] = 0;
    const i32x4 z4 = {0, 0, 0, 0};
#if __clang_major__ >= 23
    const i32x8 z8 = {0, 0, 0, 0, 0, 0, 0, 0};
    __builtin_amdgcn_tensor_load_to_lds(g0, g1, z4, z4, z8, 0);
#else
    __builtin_amdgcn_tensor_load_to_lds(g0, g1, z4, z4, 0);
#endif
}

// ---------------------------------------------------------------------------
// bf16 GEMM:  val = act( alpha*A.Bt^T + biasVec + addMat + resid )
//   A : bf16 [M,K] row-major (batched: zo*aOuter + zi*aInner)
//   Bt: bf16 [N,K] row-major (pre-transposed B)
// Outputs: outF (fp32, nullable) and/or outH (bf16, nullable; vtMode=1 stores
// into [B,H,DH,S] transposed V layout). Grid: x=N/BN, y=M/BM, z=batch.
// 256 threads = 8 waves (WAVES_M x WAVES_N); wave 0 drives the TDM.
// ---------------------------------------------------------------------------
template <int BM, int BN, int BK, int WAVES_M, int WAVES_N>
__global__ __launch_bounds__(256, 1) void gemm_bf16t(
    const us* __restrict__ A,  ll aOuter, ll aInner, int lda,
    const us* __restrict__ Bt, ll bOuter, ll bInner, int ldb,
    float*    __restrict__ outF, ll cOuter, ll cInner, int ldc,
    us*       __restrict__ outH, ll hOuter, ll hInner, int ldh, int vtMode,
    const float* __restrict__ biasVec,
    const float* __restrict__ addMat, ll dOuter, ll dInner, int ldd,
    const float* __restrict__ resid, int ldr,
    int K, int Hdim, float alpha, int act)
{
    static_assert(BK == 32, "TDM descriptor hardcodes 32-elem K tiles");
    constexpr int AP   = BK + 8;                     // u16 pitch (matches TDM pad)
    constexpr int WM_T = BM / (WAVES_M * 16);
    constexpr int WN_T = BN / (WAVES_N * 16);

    __shared__ us As[2][BM * AP];
    __shared__ us Bs[2][BN * AP];

    const int z  = blockIdx.z;
    const int zo = z / Hdim, zi = z % Hdim;
    const int m0 = blockIdx.y * BM;
    const int n0 = blockIdx.x * BN;

    const us* Ab = A  + (ll)zo * aOuter + (ll)zi * aInner + (ll)m0 * lda;
    const us* Bb = Bt + (ll)zo * bOuter + (ll)zi * bInner + (ll)n0 * ldb;

    const int lane   = threadIdx.x & 31;
    const int waveM  = (threadIdx.x >> 5) / WAVES_N;
    const int waveN  = (threadIdx.x >> 5) % WAVES_N;
    const int lrow   = lane & 15;
    const int khalfA = (lane & 16) ? 8 : 0;
    const int kbaseB = (lane & 16) ? 8 : 0;
    const bool issuer = (threadIdx.x < 32);          // wave 0 drives TDM

    const unsigned ldsA0 = (unsigned)(size_t)&As[0][0];
    const unsigned ldsA1 = (unsigned)(size_t)&As[1][0];
    const unsigned ldsB0 = (unsigned)(size_t)&Bs[0][0];
    const unsigned ldsB1 = (unsigned)(size_t)&Bs[1][0];

    v8f acc[WM_T][WN_T] = {};

    const int KT = K / BK;
    if (issuer) {
        tdm_load_tile_2d(ldsA0, Ab, BM, lda);
        tdm_load_tile_2d(ldsB0, Bb, BN, ldb);
    }
    for (int kt = 0; kt < KT; ++kt) {
        const int cur = kt & 1;
        __syncthreads();   // all waves done reading buf[cur^1] (previous iter)
        if (issuer) {
            if (kt + 1 < KT) {
                const int ko = (kt + 1) * BK;
                tdm_load_tile_2d(cur ? ldsA0 : ldsA1, Ab + ko, BM, lda);
                tdm_load_tile_2d(cur ? ldsB0 : ldsB1, Bb + ko, BN, ldb);
                __builtin_amdgcn_s_wait_tensorcnt(2);  // buf[cur]'s 2 loads done
            } else {
                __builtin_amdgcn_s_wait_tensorcnt(0);
            }
        }
        __syncthreads();   // publish buf[cur] to all waves

        v16bf af[WM_T], bfr[WN_T];
        #pragma unroll
        for (int i = 0; i < WM_T; ++i)
            af[i] = load_frag_A(&As[cur][(waveM * WM_T * 16 + i * 16 + lrow) * AP + khalfA]);
        #pragma unroll
        for (int j = 0; j < WN_T; ++j)
            bfr[j] = load_frag_B(&Bs[cur][(waveN * WN_T * 16 + j * 16 + lrow) * AP + kbaseB]);
        #pragma unroll
        for (int i = 0; i < WM_T; ++i)
            #pragma unroll
            for (int j = 0; j < WN_T; ++j)
                acc[i][j] = __builtin_amdgcn_wmma_f32_16x16x32_bf16(
                    false, af[i], false, bfr[j], (short)0, acc[i][j], false, false);
    }

    // ---- epilogue ---------------------------------------------------------
    float* Cf = outF ? outF + (ll)zo * cOuter + (ll)zi * cInner : nullptr;
    us*    Ch = outH ? outH + (ll)zo * hOuter + (ll)zi * hInner : nullptr;
    const float* Db = addMat ? addMat + (ll)zo * dOuter + (ll)zi * dInner : nullptr;

    #pragma unroll
    for (int i = 0; i < WM_T; ++i) {
        #pragma unroll
        for (int j = 0; j < WN_T; ++j) {
            const int col   = n0 + (waveN * WN_T + j) * 16 + (lane & 15);
            const int rbase = m0 + (waveM * WM_T + i) * 16 + ((lane & 16) ? 8 : 0);
            const float bvv = biasVec ? biasVec[col] : 0.0f;
            #pragma unroll
            for (int r = 0; r < 8; ++r) {
                const int row = rbase + r;
                float val = acc[i][j][r] * alpha + bvv;
                if (Db)    val += Db[(ll)row * ldd + col];
                if (resid) val += resid[(ll)row * ldr + col];
                if (act == 1) val = 0.5f * val * (1.0f + erff(val * 0.70710678118654752f));
                if (Cf) Cf[(ll)row * ldc + col] = val;
                if (Ch) {
                    if (vtMode) {  // V -> [B,H,DH,S] bf16
                        const int b = row >> 9, s = row & 511;
                        const int h = col >> 6, d = col & 63;
                        Ch[((((ll)b * Hh + h) * DHd + d) << 9) + s] = f2bf(val);
                    } else {
                        Ch[(ll)row * ldh + col] = f2bf(val);
                    }
                }
            }
        }
    }
}

// ---------------------------------------------------------------------------
// fp32 [K,N] -> bf16 [N,K] transpose-convert (per z: one layer's weight)
// ---------------------------------------------------------------------------
__global__ __launch_bounds__(256) void transpose_bf16_kernel(
    const float* __restrict__ in, us* __restrict__ out,
    int K, int N, ll inStride, ll outStride)
{
    __shared__ us t[32 * 33];
    const float* ip = in + (ll)blockIdx.z * inStride;
    us* op = out + (ll)blockIdx.z * outStride;
    const int k0 = blockIdx.y * 32, n0 = blockIdx.x * 32;
    #pragma unroll
    for (int it = 0; it < 4; ++it) {
        int idx = threadIdx.x + it * 256;
        int kr = idx >> 5, nc = idx & 31;
        t[kr * 33 + nc] = f2bf(ip[(ll)(k0 + kr) * N + n0 + nc]);
    }
    __syncthreads();
    #pragma unroll
    for (int it = 0; it < 4; ++it) {
        int idx = threadIdx.x + it * 256;
        int nr = idx >> 5, kc = idx & 31;
        op[(ll)(n0 + nr) * K + k0 + kc] = t[kc * 33 + nr];
    }
}

// fp32 -> (fp32 copy, bf16 copy), float4-wide
__global__ __launch_bounds__(256) void convert_kernel(
    const float* __restrict__ in, float* __restrict__ outF, us* __restrict__ outH)
{
    const ll i = ((ll)blockIdx.x * 256 + threadIdx.x) * 4;
    float4 f = *(const float4*)(in + i);
    *(float4*)(outF + i) = f;
    us* h = outH + i;
    h[0] = f2bf(f.x); h[1] = f2bf(f.y); h[2] = f2bf(f.z); h[3] = f2bf(f.w);
}

// ---------------------------------------------------------------------------
DEV int rel_bucket_dev(int rp, int num_buckets, int max_distance) {
    const int nb  = num_buckets >> 1;
    const int ret = (rp > 0) ? nb : 0;
    int n = rp < 0 ? -rp : rp;
    const int me = nb >> 1;
    if (n < me) return ret + n;
    float t = logf((float)n / (float)me) /
              logf((float)max_distance / (float)me) * (float)(nb - me);
    int vi = me + (int)t;
    vi = vi < (nb - 1) ? vi : (nb - 1);
    return ret + vi;
}

__global__ __launch_bounds__(256) void bias_kernel(
    const int* __restrict__ pos, const int* __restrict__ bbox,
    const float* __restrict__ mask,
    const float* __restrict__ rel_w, const float* __restrict__ relx_w,
    const float* __restrict__ rely_w, float* __restrict__ bias)
{
    const int kcol = blockIdx.x * blockDim.x + threadIdx.x;
    const int qrow = blockIdx.y;
    const int b    = blockIdx.z;

    const int pq = pos[b * Ss + qrow], pk = pos[b * Ss + kcol];
    const int b1 = rel_bucket_dev(pk - pq, 32, 128);
    const int pxq = bbox[(b * Ss + qrow) * 4 + 0], pxk = bbox[(b * Ss + kcol) * 4 + 0];
    const int pyq = bbox[(b * Ss + qrow) * 4 + 3], pyk = bbox[(b * Ss + kcol) * 4 + 3];
    const int bx = rel_bucket_dev(pxk - pxq, 64, 256);
    const int by = rel_bucket_dev(pyk - pyq, 64, 256);
    const float m = mask[b * Ss + kcol];
    const float inv_sqrt = 0.125f;

    #pragma unroll
    for (int h = 0; h < Hh; ++h) {
        float w = rel_w[b1 * Hh + h] + relx_w[bx * Hh + h] + rely_w[by * Hh + h];
        bias[(((ll)(b * Hh + h)) * Ss + qrow) * Ss + kcol] = w * inv_sqrt + m;
    }
}

// ---------------------------------------------------------------------------
// Row softmax (cogview == standard max-shifted softmax); fp32 in, bf16 out.
// ---------------------------------------------------------------------------
__global__ __launch_bounds__(256) void softmax_kernel(
    const float* __restrict__ s, us* __restrict__ p16)
{
    const ll row = (ll)blockIdx.x * 8 + (threadIdx.x >> 5);
    const int lane = threadIdx.x & 31;
    const float* p = s + row * Ss;
    us* o = p16 + row * Ss;

    float vals[16];
    float mx = -INFINITY;
    #pragma unroll
    for (int i = 0; i < 16; ++i) { vals[i] = p[lane + i * 32]; mx = fmaxf(mx, vals[i]); }
    mx = wave_max(mx);
    float sum = 0.f;
    #pragma unroll
    for (int i = 0; i < 16; ++i) { vals[i] = expf(vals[i] - mx); sum += vals[i]; }
    sum = wave_sum(sum);
    const float inv = 1.0f / sum;
    #pragma unroll
    for (int i = 0; i < 16; ++i) o[lane + i * 32] = f2bf(vals[i] * inv);
}

// ---------------------------------------------------------------------------
// LayerNorm over D=768; writes fp32 and (optional) bf16. 1 wave / row.
// ---------------------------------------------------------------------------
__global__ __launch_bounds__(256) void layernorm_kernel(
    const float* __restrict__ in, const float* __restrict__ g,
    const float* __restrict__ b,  float* __restrict__ out, us* __restrict__ out16)
{
    const ll row = (ll)blockIdx.x * 8 + (threadIdx.x >> 5);
    const int lane = threadIdx.x & 31;
    const float* p = in + row * Dm;

    float v[24];
    float s = 0.f, s2 = 0.f;
    #pragma unroll
    for (int i = 0; i < 24; ++i) {
        v[i] = p[lane + i * 32];
        s += v[i]; s2 += v[i] * v[i];
    }
    s  = wave_sum(s);
    s2 = wave_sum(s2);
    const float mu  = s * (1.0f / Dm);
    const float var = s2 * (1.0f / Dm) - mu * mu;
    const float rs  = rsqrtf(var + 1e-5f);
    #pragma unroll
    for (int i = 0; i < 24; ++i) {
        const int c = lane + i * 32;
        const float y = (v[i] - mu) * rs * g[c] + b[c];
        out[row * Dm + c] = y;
        if (out16) out16[row * Dm + c] = f2bf(y);
    }
}

// ---------------------------------------------------------------------------
extern "C" void kernel_launch(void* const* d_in, const int* in_sizes, int n_in,
                              void* d_out, int out_size, void* d_ws, size_t ws_size,
                              hipStream_t stream)
{
    (void)in_sizes; (void)n_in; (void)out_size; (void)ws_size;

    const float* hidden = (const float*)d_in[0];
    const float* mask   = (const float*)d_in[1];
    const int*   pos    = (const int*)d_in[2];
    const int*   bbox   = (const int*)d_in[3];
    const float* Wq  = (const float*)d_in[4];
    const float* bq  = (const float*)d_in[5];
    const float* Wk  = (const float*)d_in[6];
    const float* bk  = (const float*)d_in[7];
    const float* Wv  = (const float*)d_in[8];
    const float* bv  = (const float*)d_in[9];
    const float* Wo  = (const float*)d_in[10];
    const float* bo  = (const float*)d_in[11];
    const float* ln1g = (const float*)d_in[12];
    const float* ln1b = (const float*)d_in[13];
    const float* Wi  = (const float*)d_in[14];
    const float* bi  = (const float*)d_in[15];
    const float* Wo2 = (const float*)d_in[16];
    const float* bo2 = (const float*)d_in[17];
    const float* ln2g = (const float*)d_in[18];
    const float* ln2b = (const float*)d_in[19];
    const float* rel_w  = (const float*)d_in[20];
    const float* relx_w = (const float*)d_in[21];
    const float* rely_w = (const float*)d_in[22];
    float* out = (float*)d_out;

    // ---- workspace carve --------------------------------------------------
    char* wsp = (char*)d_ws;
    auto allocF = [&](size_t n) { float* p = (float*)wsp; wsp += n * sizeof(float); return p; };
    auto allocH = [&](size_t n) { us* p = (us*)wsp; wsp += n * sizeof(us); return p; };
    const size_t SCORES = (size_t)Bb * Hh * Ss * Ss;    // 12.6M
    const size_t ACT    = (size_t)Mrows * Dm;           // 1.57M
    const size_t WDD    = (size_t)Lnum * Dm * Dm;       // 7.08M
    const size_t WDF    = (size_t)Lnum * Dm * FFd;      // 28.3M

    float* bias   = allocF(SCORES);
    float* scores = allocF(SCORES);
    float* x      = allocF(ACT);        // fp32 residual stream
    float* tmp    = allocF(ACT);        // pre-LN
    float* attn   = allocF(ACT);        // fp32 residual (post-LN1)
    us* x16    = allocH(ACT);
    us* q16    = allocH(ACT);
    us* k16    = allocH(ACT);
    us* vt16   = allocH(ACT);           // [B,H,DH,S]
    us* ctx16  = allocH(ACT);
    us* attn16 = allocH(ACT);
    us* p16    = allocH(SCORES);
    us* ffh16  = allocH((size_t)Mrows * FFd);
    us* wqT = allocH(WDD);
    us* wkT = allocH(WDD);
    us* wvT = allocH(WDD);
    us* woT = allocH(WDD);
    us* wiT = allocH(WDF);              // [FF,D] per layer
    us* wo2T = allocH(WDF);             // [D,FF] per layer

    const dim3 blk(256);

    // weights -> bf16 [N][K], once per forward
    transpose_bf16_kernel<<<dim3(Dm/32, Dm/32, Lnum), blk, 0, stream>>>(
        Wq, wqT, Dm, Dm, (ll)Dm*Dm, (ll)Dm*Dm);
    transpose_bf16_kernel<<<dim3(Dm/32, Dm/32, Lnum), blk, 0, stream>>>(
        Wk, wkT, Dm, Dm, (ll)Dm*Dm, (ll)Dm*Dm);
    transpose_bf16_kernel<<<dim3(Dm/32, Dm/32, Lnum), blk, 0, stream>>>(
        Wv, wvT, Dm, Dm, (ll)Dm*Dm, (ll)Dm*Dm);
    transpose_bf16_kernel<<<dim3(Dm/32, Dm/32, Lnum), blk, 0, stream>>>(
        Wo, woT, Dm, Dm, (ll)Dm*Dm, (ll)Dm*Dm);
    transpose_bf16_kernel<<<dim3(FFd/32, Dm/32, Lnum), blk, 0, stream>>>(
        Wi, wiT, Dm, FFd, (ll)Dm*FFd, (ll)Dm*FFd);
    transpose_bf16_kernel<<<dim3(Dm/32, FFd/32, Lnum), blk, 0, stream>>>(
        Wo2, wo2T, FFd, Dm, (ll)Dm*FFd, (ll)Dm*FFd);

    // x / x16 <- hidden_states
    convert_kernel<<<ACT / 1024, blk, 0, stream>>>(hidden, x, x16);

    // attention bias
    bias_kernel<<<dim3(Ss / 256, Ss, Bb), 256, 0, stream>>>(
        pos, bbox, mask, rel_w, relx_w, rely_w, bias);

    const dim3 gProj(Dm / 128, Mrows / 128, 1);        // (6,16)
    const dim3 gFF1(FFd / 128, Mrows / 128, 1);        // (24,16)
    const dim3 gScore(Ss / 128, Ss / 128, Bb * Hh);    // (4,4,48)
    const dim3 gCtx(DHd / 64, Ss / 128, Bb * Hh);      // (1,4,48)
    const ll SD  = (ll)Ss * Dm;
    const ll SS2 = (ll)Ss * Ss;
    const ll VH  = (ll)Hh * DHd * Ss;                  // vt16 per-batch stride
    const ll VD  = (ll)DHd * Ss;                       // vt16 per-head stride

    for (int i = 0; i < Lnum; ++i) {
        const us* wq = wqT + (ll)i * Dm * Dm;
        const us* wk = wkT + (ll)i * Dm * Dm;
        const us* wv = wvT + (ll)i * Dm * Dm;
        const us* wo = woT + (ll)i * Dm * Dm;
        const us* wi = wiT + (ll)i * Dm * FFd;
        const us* wo2 = wo2T + (ll)i * Dm * FFd;

        // Q, K: bf16 out only
        gemm_bf16t<128, 128, 32, 4, 2><<<gProj, blk, 0, stream>>>(
            x16, 0, 0, Dm, wq, 0, 0, Dm,
            nullptr, 0, 0, 0, q16, 0, 0, Dm, 0,
            bq + i * Dm, nullptr, 0, 0, 0, nullptr, 0, Dm, 1, 1.0f, 0);
        gemm_bf16t<128, 128, 32, 4, 2><<<gProj, blk, 0, stream>>>(
            x16, 0, 0, Dm, wk, 0, 0, Dm,
            nullptr, 0, 0, 0, k16, 0, 0, Dm, 0,
            bk + i * Dm, nullptr, 0, 0, 0, nullptr, 0, Dm, 1, 1.0f, 0);
        // V: bf16 out in transposed [B,H,DH,S] layout
        gemm_bf16t<128, 128, 32, 4, 2><<<gProj, blk, 0, stream>>>(
            x16, 0, 0, Dm, wv, 0, 0, Dm,
            nullptr, 0, 0, 0, vt16, 0, 0, 0, 1,
            bv + i * Dm, nullptr, 0, 0, 0, nullptr, 0, Dm, 1, 1.0f, 0);

        // scores = q.k^T/8 + bias (fp32) per (b,h)
        gemm_bf16t<128, 128, 32, 4, 2><<<gScore, blk, 0, stream>>>(
            q16, SD, DHd, Dm, k16, SD, DHd, Dm,
            scores, (ll)Hh * SS2, SS2, Ss, nullptr, 0, 0, 0, 0,
            nullptr, bias, (ll)Hh * SS2, SS2, Ss,
            nullptr, 0, DHd, Hh, 0.125f, 0);

        softmax_kernel<<<(Bb * Hh * Ss) / 8, blk, 0, stream>>>(scores, p16);

        // ctx = probs.v per (b,h) -> bf16 [B,S,D]
        gemm_bf16t<128, 64, 32, 4, 2><<<gCtx, blk, 0, stream>>>(
            p16, (ll)Hh * SS2, SS2, Ss, vt16, VH, VD, Ss,
            nullptr, 0, 0, 0, ctx16, SD, DHd, Dm, 0,
            nullptr, nullptr, 0, 0, 0, nullptr, 0, Ss, Hh, 1.0f, 0);

        // tmp = ctx.Wo + bo + x (fp32)
        gemm_bf16t<128, 128, 32, 4, 2><<<gProj, blk, 0, stream>>>(
            ctx16, 0, 0, Dm, wo, 0, 0, Dm,
            tmp, 0, 0, Dm, nullptr, 0, 0, 0, 0,
            bo + i * Dm, nullptr, 0, 0, 0, x, Dm, Dm, 1, 1.0f, 0);
        layernorm_kernel<<<Mrows / 8, blk, 0, stream>>>(
            tmp, ln1g + i * Dm, ln1b + i * Dm, attn, attn16);

        // ffh16 = gelu(attn.Wi + bi), bf16 only
        gemm_bf16t<128, 128, 32, 4, 2><<<gFF1, blk, 0, stream>>>(
            attn16, 0, 0, Dm, wi, 0, 0, Dm,
            nullptr, 0, 0, 0, ffh16, 0, 0, FFd, 0,
            bi + i * FFd, nullptr, 0, 0, 0, nullptr, 0, Dm, 1, 1.0f, 1);

        // tmp = ffh.Wo2 + bo2 + attn (fp32)
        gemm_bf16t<128, 128, 32, 4, 2><<<gProj, blk, 0, stream>>>(
            ffh16, 0, 0, FFd, wo2, 0, 0, FFd,
            tmp, 0, 0, Dm, nullptr, 0, 0, 0, 0,
            bo2 + i * Dm, nullptr, 0, 0, 0, attn, Dm, FFd, 1, 1.0f, 0);
        layernorm_kernel<<<Mrows / 8, blk, 0, stream>>>(
            tmp, ln2g + i * Dm, ln2b + i * Dm, (i == Lnum - 1) ? out : x, x16);
    }
}
